// Block_29343216566701
// MI455X (gfx1250) — matrix operand order repaired and checked by reading, not compiled, not run on hardware
//
#include <hip/hip_runtime.h>
#include <hip/hip_bf16.h>
#include <math.h>

// ---------------- types ----------------
typedef __attribute__((ext_vector_type(16))) __bf16          v16bf;
typedef __attribute__((ext_vector_type(8)))  float           v8f;
typedef __attribute__((ext_vector_type(8)))  unsigned short  us8;
typedef __attribute__((ext_vector_type(4)))  int             i32x4;
typedef __attribute__((address_space(1)))    i32x4           gi32x4;  // global int4
typedef __attribute__((address_space(3)))    i32x4           li32x4;  // LDS int4

#define N_EMBD 768
#define TSEQ   1024
#define BATCH  8
#define NHEAD  12
#define HD     64
#define QKVC   2304   // 3*N_EMBD
#define FFC    3072   // 4*N_EMBD

// fp32 -> bf16 round-to-nearest-even
__device__ __forceinline__ unsigned short f2bf(float f) {
  unsigned int u = __float_as_uint(f);
  u += 0x7fffu + ((u >> 16) & 1u);
  return (unsigned short)(u >> 16);
}

__device__ __forceinline__ v8f zero8() {
  v8f z;
#pragma unroll
  for (int i = 0; i < 8; i++) z[i] = 0.0f;
  return z;
}

// ---------------- CDNA5 async global->LDS copy (ASYNCcnt path), guarded ----------------
#if defined(__gfx1250__) && __has_builtin(__builtin_amdgcn_global_load_async_to_lds_b128)
#define HAVE_ASYNC_LDS 1
#else
#define HAVE_ASYNC_LDS 0
#endif

__device__ __forceinline__ void async_copy_b128(void* lds, const void* gsrc) {
#if HAVE_ASYNC_LDS
  __builtin_amdgcn_global_load_async_to_lds_b128(
      (gi32x4*)gsrc, (li32x4*)lds, 0, 0);
#else
  *(uint4*)lds = *(const uint4*)gsrc;
#endif
}
#define WAIT_ASYNC(n) asm volatile("s_wait_asynccnt " #n ::: "memory")

// A-matrix (16x32 bf16) per-lane load from a row-major [16][ld] bf16 tile.
// ISA layout: lane L row = L&15; halves 0..7 -> K = 8*(L>>4)+j ; halves 8..15 -> K = 16+8*(L>>4)+j
__device__ __forceinline__ v16bf load_A16(const unsigned short* base, int ld, int lane) {
  const int r = lane & 15, g = lane >> 4;
  const unsigned short* p = base + r * ld + 8 * g;
  union { us8 u[2]; v16bf v; } t;
  t.u[0] = *(const us8*)p;
  t.u[1] = *(const us8*)(p + 16);
  return t.v;
}

// B-matrix (32x16 bf16) per-lane load from a [16 cols][ld along K] tile (B stored transposed).
// ISA layout: lane L col = L&15; halves 0..7 -> K = 8*(L>>4)+j ; halves 8..15 -> K = 16+8*(L>>4)+j
__device__ __forceinline__ v16bf load_B16(const unsigned short* base, int ld, int lane) {
  const int c = lane & 15, g = lane >> 4;
  const unsigned short* p = base + c * ld + 8 * g;
  union { us8 u[2]; v16bf v; } t;
  t.u[0] = *(const us8*)p;
  t.u[1] = *(const us8*)(p + 16);
  return t.v;
}

__device__ __forceinline__ v8f wmma_bf16(v16bf a, v16bf b, v8f c) {
  return __builtin_amdgcn_wmma_f32_16x16x32_bf16(false, a, false, b, (short)0, c, false, false);
}

// ---------------- fp32 W[K][N] -> bf16 WT[N][K] (so GEMM B staging is contiguous) ----------------
__global__ __launch_bounds__(256) void cvt_transpose_kernel(const float* __restrict__ W,
                                                            unsigned short* __restrict__ WT,
                                                            int K, int N) {
  int idx = blockIdx.x * 256 + threadIdx.x;
  if (idx < K * N) {
    int k = idx / N, n = idx - k * N;
    WT[(size_t)n * K + k] = f2bf(W[idx]);
  }
}

// ---------------- LayerNorm (fp32 in, bf16 out), one block per row of 768 ----------------
__global__ __launch_bounds__(256) void ln_bf16_kernel(const float* __restrict__ x,
                                                      const float* __restrict__ gamma,
                                                      const float* __restrict__ beta,
                                                      unsigned short* __restrict__ out) {
  __shared__ float sred[256];
  const int row = blockIdx.x, tid = threadIdx.x;
  const float* xr = x + (size_t)row * N_EMBD;
  float v0 = xr[tid], v1 = xr[tid + 256], v2 = xr[tid + 512];
  sred[tid] = v0 + v1 + v2;
  __syncthreads();
  for (int o = 128; o > 0; o >>= 1) { if (tid < o) sred[tid] += sred[tid + o]; __syncthreads(); }
  float mu = sred[0] * (1.0f / N_EMBD);
  __syncthreads();
  float d0 = v0 - mu, d1 = v1 - mu, d2 = v2 - mu;
  sred[tid] = d0 * d0 + d1 * d1 + d2 * d2;
  __syncthreads();
  for (int o = 128; o > 0; o >>= 1) { if (tid < o) sred[tid] += sred[tid + o]; __syncthreads(); }
  float rstd = rsqrtf(sred[0] * (1.0f / N_EMBD) + 1e-5f);
  unsigned short* orow = out + (size_t)row * N_EMBD;
  orow[tid]       = f2bf(d0 * rstd * gamma[tid]       + beta[tid]);
  orow[tid + 256] = f2bf(d1 * rstd * gamma[tid + 256] + beta[tid + 256]);
  orow[tid + 512] = f2bf(d2 * rstd * gamma[tid + 512] + beta[tid + 512]);
}

// ---------------- Tiled bf16 WMMA GEMM with double-buffered async LDS pipeline ----------------
// C[M,N] = A[M,K] @ W[K,N] + bias;  W passed pre-transposed as WT[N][K] bf16.
// Block: 256 threads (8 waves). Tile M=256 (8 waves x 32 rows), N=64, K step 32.
// Per K step each wave: 2 A tiles x 4 B tiles = 8 WMMAs; each B LDS load feeds 2 WMMAs.
// mode 0: out bf16 = v ; mode 1: out bf16 = gelu(v) ; mode 2: out f32 = resid + v
#define BMG 256
#define BNG 64
#define BKG 32
__global__ __launch_bounds__(256) void gemm_bf16_kernel(
    const unsigned short* __restrict__ A, const unsigned short* __restrict__ WT,
    const float* __restrict__ bias, const float* __restrict__ resid,
    float* __restrict__ outF, unsigned short* __restrict__ outB,
    int M, int N, int K, int mode) {
  __shared__ __align__(16) unsigned short As[2][BMG][BKG];
  __shared__ __align__(16) unsigned short Ws[2][BNG][BKG];
  const int tid = threadIdx.x;
  const int wid = tid >> 5;
  const int lane = tid & 31;
  const int g = lane >> 4;
  const int nn = lane & 15;
  const int row0 = blockIdx.y * BMG;
  const int col0 = blockIdx.x * BNG;

  // per-chunk staging: thread t copies A row (row0+t), 32 halves = 4x b128; plus 1x b128 of WT.
  auto issue = [&](int kc, int buf) {
    const int k0 = kc * BKG;
    const unsigned short* asrc = A + (size_t)(row0 + tid) * K + k0;
#pragma unroll
    for (int j = 0; j < 4; j++) async_copy_b128(&As[buf][tid][8 * j], asrc + 8 * j);
    const int colc = tid >> 2, part = tid & 3;
    async_copy_b128(&Ws[buf][colc][part * 8],
                    WT + (size_t)(col0 + colc) * K + k0 + part * 8);
  };

  v8f acc0[4], acc1[4];
#pragma unroll
  for (int i = 0; i < 4; i++) { acc0[i] = zero8(); acc1[i] = zero8(); }

  const int nk = K / BKG;
  issue(0, 0);
  for (int kc = 0; kc < nk; kc++) {
    const int cb = kc & 1;
    if (kc + 1 < nk) {       // prefetch next chunk into other buffer, then drain older batch
      issue(kc + 1, cb ^ 1);
      WAIT_ASYNC(0x5);       // 5 newer copies may remain in flight; chunk kc is complete
    } else {
      WAIT_ASYNC(0x0);
    }
    __syncthreads();
    v16bf a0 = load_A16(&As[cb][wid * 32][0],      BKG, lane);
    v16bf a1 = load_A16(&As[cb][wid * 32 + 16][0], BKG, lane);
#pragma unroll
    for (int nt = 0; nt < 4; nt++) {
      v16bf bm = load_B16(&Ws[cb][nt * 16][0], BKG, lane);
      acc0[nt] = wmma_bf16(a0, bm, acc0[nt]);
      acc1[nt] = wmma_bf16(a1, bm, acc1[nt]);
    }
    __syncthreads();
  }

  // epilogue (D layout: lane&15 = col, VGPR r = row r + 8*(lane>>4)); two 16-row groups per wave
#pragma unroll
  for (int sub = 0; sub < 2; sub++) {
#pragma unroll
    for (int nt = 0; nt < 4; nt++) {
      int col = col0 + nt * 16 + nn;
      float bv = bias[col];
#pragma unroll
      for (int r = 0; r < 8; r++) {
        int row = row0 + wid * 32 + sub * 16 + r + 8 * g;
        float v = (sub ? acc1[nt][r] : acc0[nt][r]) + bv;
        size_t idx = (size_t)row * N + col;
        if (mode == 0) {
          outB[idx] = f2bf(v);
        } else if (mode == 1) {
          outB[idx] = f2bf(0.5f * v * (1.0f + erff(v * 0.70710678118654752f)));
        } else {
          outF[idx] = resid[idx] + v;
        }
      }
    }
  }
}

// ---------------- Causal flash attention over bf16 qkv ----------------
// 1 block (128 thr, 4 waves) = 64 queries of one (batch, head). Wave w owns queries q0+16w..+15.
// Q and K tiles stream in via async-to-LDS; V is staged transposed (manual scatter).
__global__ __launch_bounds__(128) void attn_kernel(const unsigned short* __restrict__ qkv,
                                                   unsigned short* __restrict__ out) {
  __shared__ __align__(16) unsigned short Qs[64][64];
  __shared__ __align__(16) unsigned short Ks[32][64];
  __shared__ __align__(16) unsigned short VT[64][32];     // V transposed: [hd][key]
  __shared__ __align__(16) unsigned short Pst[4][16][32]; // per-wave P staging

  const int tid = threadIdx.x;
  const int wid = tid >> 5;
  const int lane = tid & 31;
  const int g = lane >> 4;
  const int nn = lane & 15;

  const int blk = blockIdx.x;
  const int qt = blk & 15;
  const int h  = (blk >> 4) % NHEAD;
  const int b  = blk / (16 * NHEAD);
  const int q0 = qt * 64;

  // stage Q tile: 64 rows x 64 halves (async)
#pragma unroll
  for (int rep = 0; rep < 4; rep++) {
    int idx = tid + rep * 128;
    int r = idx >> 3, c = (idx & 7) * 8;
    async_copy_b128(&Qs[r][c],
                    qkv + ((size_t)(b * TSEQ + q0 + r)) * QKVC + h * HD + c);
  }
  WAIT_ASYNC(0x0);
  __syncthreads();

  v16bf qa0 = load_A16(&Qs[wid * 16][0],  64, lane);  // hd 0..31
  v16bf qa1 = load_A16(&Qs[wid * 16][32], 64, lane);  // hd 32..63

  v8f acc[4];
#pragma unroll
  for (int i = 0; i < 4; i++) acc[i] = zero8();
  float mrow[8], lrow[8];
#pragma unroll
  for (int r = 0; r < 8; r++) { mrow[r] = -1e30f; lrow[r] = 0.0f; }

  const int qend = q0 + wid * 16 + 15;
  const int nch = (q0 + 64) >> 5;
  for (int kc = 0; kc < nch; kc++) {
    const int kbase = kc * 32;
    // stage K chunk (async, row-major) and V chunk (transposed, manual)
#pragma unroll
    for (int rep = 0; rep < 2; rep++) {
      int idx = tid + rep * 128;
      int i = idx >> 3, c = (idx & 7) * 8;
      async_copy_b128(&Ks[i][c],
                      qkv + ((size_t)(b * TSEQ + kbase + i)) * QKVC + N_EMBD + h * HD + c);
      const unsigned short* vsrc =
          qkv + ((size_t)(b * TSEQ + kbase + i)) * QKVC + 2 * N_EMBD + h * HD + c;
      us8 vd = *(const us8*)vsrc;
#pragma unroll
      for (int j = 0; j < 8; j++) VT[c + j][i] = vd[j];
    }
    WAIT_ASYNC(0x0);
    __syncthreads();

    if (kbase <= qend) {  // wave-uniform predicate: EXEC stays all-ones for WMMA
      v8f s0 = zero8(), s1 = zero8();
      s0 = wmma_bf16(qa0, load_B16(&Ks[0][0],   64, lane), s0);
      s0 = wmma_bf16(qa1, load_B16(&Ks[0][32],  64, lane), s0);
      s1 = wmma_bf16(qa0, load_B16(&Ks[16][0],  64, lane), s1);
      s1 = wmma_bf16(qa1, load_B16(&Ks[16][32], 64, lane), s1);

#pragma unroll
      for (int r = 0; r < 8; r++) {
        int rowq = q0 + wid * 16 + r + 8 * g;
        float x0 = s0[r] * 0.125f; if (kbase + nn > rowq)      x0 = -1e30f;
        float x1 = s1[r] * 0.125f; if (kbase + 16 + nn > rowq) x1 = -1e30f;
        float m = fmaxf(x0, x1);
#pragma unroll
        for (int off = 8; off > 0; off >>= 1) m = fmaxf(m, __shfl_xor(m, off, 16));
        float mn = fmaxf(mrow[r], m);
        float alpha = __expf(mrow[r] - mn);
        mrow[r] = mn;
        float e0 = __expf(x0 - mn), e1 = __expf(x1 - mn);
        float ps = e0 + e1;
#pragma unroll
        for (int off = 8; off > 0; off >>= 1) ps += __shfl_xor(ps, off, 16);
        lrow[r] = lrow[r] * alpha + ps;
#pragma unroll
        for (int nt = 0; nt < 4; nt++) acc[nt][r] *= alpha;
        Pst[wid][r + 8 * g][nn]      = f2bf(e0);
        Pst[wid][r + 8 * g][nn + 16] = f2bf(e1);
      }
      asm volatile("s_wait_dscnt 0x0" ::: "memory");  // P store -> A-layout reload, same wave
      v16bf pa = load_A16(&Pst[wid][0][0], 32, lane);
#pragma unroll
      for (int nt = 0; nt < 4; nt++) {
        v16bf vb = load_B16(&VT[nt * 16][0], 32, lane);
        acc[nt] = wmma_bf16(pa, vb, acc[nt]);
      }
    }
    __syncthreads();
  }

  // O /= l, write bf16 into attn_out [B*T][768] at head offset
#pragma unroll
  for (int r = 0; r < 8; r++) {
    float inv = 1.0f / lrow[r];
    int row = q0 + wid * 16 + r + 8 * g;
#pragma unroll
    for (int nt = 0; nt < 4; nt++) {
      int col = h * HD + nt * 16 + nn;
      out[((size_t)(b * TSEQ + row)) * N_EMBD + col] = f2bf(acc[nt][r] * inv);
    }
  }
}

// ---------------- host-side launch ----------------
extern "C" void kernel_launch(void* const* d_in, const int* in_sizes, int n_in,
                              void* d_out, int out_size, void* d_ws, size_t ws_size,
                              hipStream_t stream) {
  (void)in_sizes; (void)n_in; (void)out_size; (void)ws_size;
  const float* x      = (const float*)d_in[0];
  const float* ln1g   = (const float*)d_in[1];
  const float* ln1b   = (const float*)d_in[2];
  const float* W_attn = (const float*)d_in[3];
  const float* b_attn = (const float*)d_in[4];
  const float* W_proj = (const float*)d_in[5];
  const float* b_proj = (const float*)d_in[6];
  const float* ln2g   = (const float*)d_in[7];
  const float* ln2b   = (const float*)d_in[8];
  const float* W_fc   = (const float*)d_in[9];
  const float* b_fc   = (const float*)d_in[10];
  const float* W_fc2  = (const float*)d_in[11];
  const float* b_fc2  = (const float*)d_in[12];
  float* out = (float*)d_out;

  const int nTok = BATCH * TSEQ;  // 8192

  char* w = (char*)d_ws;
  size_t off = 0;
  auto alloc = [&](size_t bytes) -> char* {
    char* p = w + off;
    off += (bytes + 255) & ~(size_t)255;
    return p;
  };
  unsigned short* wAttnT = (unsigned short*)alloc((size_t)N_EMBD * QKVC * 2);  // [QKVC][768]
  unsigned short* wProjT = (unsigned short*)alloc((size_t)N_EMBD * N_EMBD * 2);
  unsigned short* wFcT   = (unsigned short*)alloc((size_t)N_EMBD * FFC * 2);   // [FFC][768]
  unsigned short* wFc2T  = (unsigned short*)alloc((size_t)FFC * N_EMBD * 2);   // [768][FFC]
  unsigned short* ln1o   = (unsigned short*)alloc((size_t)nTok * N_EMBD * 2);
  unsigned short* qkv    = (unsigned short*)alloc((size_t)nTok * QKVC * 2);
  unsigned short* atto   = (unsigned short*)alloc((size_t)nTok * N_EMBD * 2);
  float*          x1     = (float*)         alloc((size_t)nTok * N_EMBD * 4);
  unsigned short* ln2o   = (unsigned short*)alloc((size_t)nTok * N_EMBD * 2);
  unsigned short* hbuf   = (unsigned short*)alloc((size_t)nTok * FFC * 2);

  auto cvt = [&](const float* s, unsigned short* d, int K, int N) {
    int n = K * N;
    cvt_transpose_kernel<<<dim3((n + 255) / 256), dim3(256), 0, stream>>>(s, d, K, N);
  };
  cvt(W_attn, wAttnT, N_EMBD, QKVC);
  cvt(W_proj, wProjT, N_EMBD, N_EMBD);
  cvt(W_fc,   wFcT,   N_EMBD, FFC);
  cvt(W_fc2,  wFc2T,  FFC,    N_EMBD);

  // ln1
  ln_bf16_kernel<<<dim3(nTok), dim3(256), 0, stream>>>(x, ln1g, ln1b, ln1o);
  // qkv = ln1 @ W_attn + b_attn  -> bf16
  gemm_bf16_kernel<<<dim3(QKVC / BNG, nTok / BMG), dim3(256), 0, stream>>>(
      ln1o, wAttnT, b_attn, nullptr, nullptr, qkv, nTok, QKVC, N_EMBD, 0);
  // causal attention -> bf16
  attn_kernel<<<dim3(BATCH * NHEAD * (TSEQ / 64)), dim3(128), 0, stream>>>(qkv, atto);
  // x1 = x + atto @ W_proj + b_proj  -> f32
  gemm_bf16_kernel<<<dim3(N_EMBD / BNG, nTok / BMG), dim3(256), 0, stream>>>(
      atto, wProjT, b_proj, x, x1, nullptr, nTok, N_EMBD, N_EMBD, 2);
  // ln2
  ln_bf16_kernel<<<dim3(nTok), dim3(256), 0, stream>>>(x1, ln2g, ln2b, ln2o);
  // h = gelu(ln2 @ W_fc + b_fc) -> bf16
  gemm_bf16_kernel<<<dim3(FFC / BNG, nTok / BMG), dim3(256), 0, stream>>>(
      ln2o, wFcT, b_fc, nullptr, nullptr, hbuf, nTok, FFC, N_EMBD, 1);
  // out = x1 + h @ W_fc2 + b_fc2 -> f32
  gemm_bf16_kernel<<<dim3(N_EMBD / BNG, nTok / BMG), dim3(256), 0, stream>>>(
      hbuf, wFc2T, b_fc2, x1, out, nullptr, nTok, N_EMBD, FFC, 2);
}
